// Disjunction_57793079935284
// MI455X (gfx1250) — matrix operand order repaired and checked by reading, not compiled, not run on hardware
//
#include <hip/hip_runtime.h>

// ---------------------------------------------------------------------------
// out = X @ W^T + 0.01*(|X| @ |W|^T - 1e-4),  X:[4096,4096] W:[4096,4096] fp32
//
// Two-pass MI455X pipeline (if ws_size >= 128MB):
//   pass 1: fp32 -> bf16 (hi + lo truncation split) into workspace  (~11us @ 23.3TB/s)
//   pass 2: bf16 WMMA GEMM; tiles staged LDS via Tensor Data Mover
//           (tensor_load_to_lds, double-buffered, s_wait_tensorcnt),
//           main += Xhi*Whi + Xhi*Wlo + Xlo*Whi ; abs += |Xhi|*|Whi| (v_and)
// Fallback (small ws): single-pass fused convert+GEMM kernel.
// ---------------------------------------------------------------------------

typedef __attribute__((ext_vector_type(16))) __bf16        v16bf;
typedef __attribute__((ext_vector_type(8)))  float         v8f;
typedef __attribute__((ext_vector_type(4)))  float         f4v;
typedef __attribute__((ext_vector_type(4)))  unsigned int  u4v;
typedef __attribute__((ext_vector_type(4)))  unsigned int  tdmG0;   // group0
typedef __attribute__((ext_vector_type(8)))  int           tdmG1;   // group1
typedef __attribute__((ext_vector_type(4)))  int           tdmG2;   // group2/3

#define NDIM 4096
#define BK   32
#define LDSW 20            // uint32 per LDS row: 16 data + 4 pad (80B pitch)
#define TILE_U32 (128 * LDSW)

#if defined(__gfx1250__) && __has_builtin(__builtin_amdgcn_tensor_load_to_lds)
#define HAS_TDM 1
#else
#define HAS_TDM 0
#endif

union Frag { v16bf v; unsigned int u[8]; };

#define WMMA_BF16(A, B, C) \
  __builtin_amdgcn_wmma_f32_16x16x32_bf16(false, (A), false, (B), (short)0, (C), false, false)

// Truncation split: sign(hi)==sign(x) so |x| ~= |hi|, making the 0.01-weighted
// abs-GEMM valid on the hi fragments alone (sign clear = v_and).
__device__ __forceinline__ void splitPack(float x0, float x1,
                                          unsigned int& hiP, unsigned int& loP) {
  unsigned int u0 = __float_as_uint(x0);
  unsigned int u1 = __float_as_uint(x1);
  unsigned int h0 = u0 & 0xFFFF0000u;
  unsigned int h1 = u1 & 0xFFFF0000u;
  float r0 = x0 - __uint_as_float(h0);
  float r1 = x1 - __uint_as_float(h1);
  hiP = (h0 >> 16) | h1;
  loP = (__float_as_uint(r0) >> 16) | (__float_as_uint(r1) & 0xFFFF0000u);
}

// A fragment (16x32 bf16): lane m = lane&15, kh = (lane>=16)?8:0;
// elems 0..7 = K[kh..kh+7], elems 8..15 = K[kh+16..kh+23]
__device__ __forceinline__ void loadFragA(const unsigned int* lds, int row, int kh, Frag& f) {
  const u4v* p0 = (const u4v*)(lds + row * LDSW + (kh >> 1));
  const u4v* p1 = (const u4v*)(lds + row * LDSW + (kh >> 1) + 8);
  *(u4v*)&f.u[0] = *p0;
  *(u4v*)&f.u[4] = *p1;
}

// B fragment (32x16 bf16): lane n = lane&15, ks = (lane>=16)?16:0;
// elems 0..7 = K[ks/2..ks/2+7], elems 8..15 = K[ks/2+16..ks/2+23]
__device__ __forceinline__ void loadFragB(const unsigned int* lds, int row, int ks, Frag& f) {
  const u4v* p = (const u4v*)(lds + row * LDSW + (ks >> 2));
  *(u4v*)&f.u[0] = p[0];
  *(u4v*)&f.u[4] = p[2];
}

// ---------------------------------------------------------------------------
// Pass 1: fp32 -> packed bf16 hi/lo (truncation split), row-major [4096][4096]
// ---------------------------------------------------------------------------
__global__ __launch_bounds__(256)
void split_convert(const float* __restrict__ in,
                   unsigned int* __restrict__ hi, unsigned int* __restrict__ lo) {
  const size_t i = (size_t)blockIdx.x * 256 + threadIdx.x;   // float4 index
  const f4v v = ((const f4v*)in)[i];
  unsigned int h0, l0, h1, l1;
  splitPack(v[0], v[1], h0, l0);
  splitPack(v[2], v[3], h1, l1);
  hi[2 * i + 0] = h0;  hi[2 * i + 1] = h1;
  lo[2 * i + 0] = l0;  lo[2 * i + 1] = l1;
}

// ---------------------------------------------------------------------------
// TDM: load one 128x32 bf16 tile (row pitch 4096 elems) into LDS with the
// 80-byte padded pitch produced by the TDM pad feature (16 dw data + 4 dw pad).
// ---------------------------------------------------------------------------
#if HAS_TDM
__device__ __forceinline__ void tdm_load_tile(unsigned int ldsOff, unsigned long long gaddr) {
  tdmG0 g0;
  g0[0] = 1u;                                            // count=1 (valid D#)
  g0[1] = ldsOff;                                        // lds_addr (bytes)
  g0[2] = (unsigned int)(gaddr & 0xFFFFFFFFull);         // global_addr lo
  g0[3] = (unsigned int)(gaddr >> 32) | (2u << 30);      // global_addr hi | type=2
  tdmG1 g1;
  g1[0] = (int)((1u << 16)      // data_size = 2 bytes
              | (1u << 20)      // pad_enable
              | (3u << 22)      // pad_interval: 16 DWORDs (=64B = one tile row)
              | (3u << 25));    // pad_amount: 4 DWORDs (=16B)  -> 80B pitch
  g1[1] = (int)((NDIM & 0xFFFF) << 16);                              // tensor_dim0 lo16
  g1[2] = (int)(((NDIM >> 16) & 0xFFFF) | ((NDIM & 0xFFFF) << 16));  // dim0 hi | dim1 lo
  g1[3] = (int)(((NDIM >> 16) & 0xFFFF) | (BK << 16));               // dim1 hi | tile_dim0=32
  g1[4] = 128;                                                       // tile_dim1=128, tile_dim2=0
  g1[5] = NDIM;                                                      // tensor_dim0_stride lo32
  g1[6] = 0;
  g1[7] = 0;
  tdmG2 gz = {0, 0, 0, 0};
#if __clang_major__ >= 23
  tdmG1 gz8 = {0, 0, 0, 0, 0, 0, 0, 0};
  __builtin_amdgcn_tensor_load_to_lds(g0, g1, gz, gz, gz8, 0);
#else
  __builtin_amdgcn_tensor_load_to_lds(g0, g1, gz, gz, 0);
#endif
}
#endif

// Fallback tile copy (bf16 already packed as uints, global pitch NDIM/2 uints)
__device__ __forceinline__ void copy_tile(unsigned int* dst, const unsigned int* src, int t) {
#pragma unroll
  for (int j = 0; j < 2; ++j) {
    const int v = t + j * 256;          // 0..511 u4v chunks
    const int row = v >> 2, c = v & 3;  // 4 u4v per 64B row
    const u4v d = *(const u4v*)(src + (size_t)row * (NDIM / 2) + c * 4);
    *(u4v*)(dst + row * LDSW + c * 4) = d;
  }
}

// ---------------------------------------------------------------------------
// Pass 2: bf16 WMMA GEMM, TDM-staged + double-buffered LDS
// ---------------------------------------------------------------------------
__global__ __launch_bounds__(256)
void gemm_bf16_tdm(const unsigned int* __restrict__ Ahi, const unsigned int* __restrict__ Alo,
                   const unsigned int* __restrict__ Bhi, const unsigned int* __restrict__ Blo,
                   float* __restrict__ Out) {
  __shared__ __align__(16) unsigned int sAhi[2][TILE_U32];
  __shared__ __align__(16) unsigned int sAlo[2][TILE_U32];
  __shared__ __align__(16) unsigned int sBhi[2][TILE_U32];
  __shared__ __align__(16) unsigned int sBlo[2][TILE_U32];

  const int t    = threadIdx.x;
  const int lane = t & 31;
  const int wave = t >> 5;
  const int wm   = wave & 3;          // 4 waves along M (32 rows each)
  const int wn   = wave >> 2;         // 2 waves along N (64 cols each)
  const int mRow = wm * 32;
  const int nRow = wn * 64;
  const int fm   = lane & 15;
  const int kh   = (lane >> 4) * 8;
  const int ks   = (lane >> 4) * 16;

  const int rowA0 = blockIdx.y * 128;
  const int rowB0 = blockIdx.x * 128;

  const v8f vzero = {0.f, 0.f, 0.f, 0.f, 0.f, 0.f, 0.f, 0.f};
  v8f accM[2][4], accA[2][4];
#pragma unroll
  for (int i = 0; i < 2; ++i)
#pragma unroll
    for (int j = 0; j < 4; ++j) { accM[i][j] = vzero; accA[i][j] = vzero; }

  auto issue = [&](int k0, int b) {
#if HAS_TDM
    if (wave == 0) {
      const unsigned long long offA = ((unsigned long long)rowA0 * NDIM + k0) * 2ull;
      const unsigned long long offB = ((unsigned long long)rowB0 * NDIM + k0) * 2ull;
      tdm_load_tile((unsigned int)(size_t)&sAhi[b][0], (unsigned long long)(size_t)Ahi + offA);
      tdm_load_tile((unsigned int)(size_t)&sAlo[b][0], (unsigned long long)(size_t)Alo + offA);
      tdm_load_tile((unsigned int)(size_t)&sBhi[b][0], (unsigned long long)(size_t)Bhi + offB);
      tdm_load_tile((unsigned int)(size_t)&sBlo[b][0], (unsigned long long)(size_t)Blo + offB);
    }
#else
    const size_t uoffA = (size_t)rowA0 * (NDIM / 2) + (k0 >> 1);
    const size_t uoffB = (size_t)rowB0 * (NDIM / 2) + (k0 >> 1);
    copy_tile(&sAhi[b][0], Ahi + uoffA, t);
    copy_tile(&sAlo[b][0], Alo + uoffA, t);
    copy_tile(&sBhi[b][0], Bhi + uoffB, t);
    copy_tile(&sBlo[b][0], Blo + uoffB, t);
#endif
  };

  issue(0, 0);  // prologue: chunk 0 -> buffer 0

  const int nChunks = NDIM / BK;
#pragma unroll 1
  for (int i = 0; i < nChunks; ++i) {
    const int b = i & 1;
#if HAS_TDM
    if (wave == 0) __builtin_amdgcn_s_wait_tensorcnt(0);   // chunk i landed
#endif
    __syncthreads();  // publish buf[b]; all waves done reading buf[1-b]
    if (i + 1 < nChunks) issue((i + 1) * BK, 1 - b);       // overlap DMA w/ WMMA

    const unsigned int* pAhi = sAhi[b];
    const unsigned int* pAlo = sAlo[b];
    const unsigned int* pBhi = sBhi[b];
    const unsigned int* pBlo = sBlo[b];

    Frag aH[2], aL[2], bH[4], bL[4], bA[4];
#pragma unroll
    for (int mi = 0; mi < 2; ++mi) {
      const int r = mRow + mi * 16 + fm;
      loadFragA(pAhi, r, kh, aH[mi]);
      loadFragA(pAlo, r, kh, aL[mi]);
    }
#pragma unroll
    for (int ni = 0; ni < 4; ++ni) {
      const int r = nRow + ni * 16 + fm;
      loadFragB(pBhi, r, ks, bH[ni]);
      loadFragB(pBlo, r, ks, bL[ni]);
#pragma unroll
      for (int j = 0; j < 8; ++j) bA[ni].u[j] = bH[ni].u[j] & 0x7FFF7FFFu;
    }
#pragma unroll
    for (int mi = 0; mi < 2; ++mi) {
      Frag aA;
#pragma unroll
      for (int j = 0; j < 8; ++j) aA.u[j] = aH[mi].u[j] & 0x7FFF7FFFu;
#pragma unroll
      for (int ni = 0; ni < 4; ++ni) {
        accM[mi][ni] = WMMA_BF16(aH[mi].v, bH[ni].v, accM[mi][ni]);
        accM[mi][ni] = WMMA_BF16(aH[mi].v, bL[ni].v, accM[mi][ni]);
        accM[mi][ni] = WMMA_BF16(aL[mi].v, bH[ni].v, accM[mi][ni]);
        accA[mi][ni] = WMMA_BF16(aA.v,     bA[ni].v, accA[mi][ni]);
      }
    }
  }

  // epilogue: out = main + 0.01*abs - 1e-6  (DELTA*bias folded)
  const int rBase = rowA0 + mRow + (lane >> 4) * 8;
  const int cBase = rowB0 + nRow + (lane & 15);
#pragma unroll
  for (int mi = 0; mi < 2; ++mi) {
#pragma unroll
    for (int ni = 0; ni < 4; ++ni) {
      float* o = Out + (size_t)(rBase + mi * 16) * NDIM + (cBase + ni * 16);
#pragma unroll
      for (int r = 0; r < 8; ++r)
        o[(size_t)r * NDIM] = accM[mi][ni][r] + 0.01f * accA[mi][ni][r] - 1.0e-6f;
    }
  }
}

// ---------------------------------------------------------------------------
// Fallback: single-pass fused convert+GEMM (round-1 kernel, no workspace)
// ---------------------------------------------------------------------------
__global__ __launch_bounds__(256)
void fused_gemm_absgemm(const float* __restrict__ X,
                        const float* __restrict__ W,
                        float* __restrict__ Out) {
  __shared__ __align__(16) unsigned int sAhi[TILE_U32];
  __shared__ __align__(16) unsigned int sAlo[TILE_U32];
  __shared__ __align__(16) unsigned int sBhi[TILE_U32];
  __shared__ __align__(16) unsigned int sBlo[TILE_U32];

  const int t    = threadIdx.x;
  const int lane = t & 31;
  const int wave = t >> 5;
  const int wm   = wave & 3;
  const int wn   = wave >> 2;
  const int mRow = wm * 32;
  const int nRow = wn * 64;
  const int fm   = lane & 15;
  const int kh   = (lane >> 4) * 8;
  const int ks   = (lane >> 4) * 16;

  const int rowA0 = blockIdx.y * 128;
  const int rowB0 = blockIdx.x * 128;

  const v8f vzero = {0.f, 0.f, 0.f, 0.f, 0.f, 0.f, 0.f, 0.f};
  v8f accM[2][4], accA[2][4];
#pragma unroll
  for (int i = 0; i < 2; ++i)
#pragma unroll
    for (int j = 0; j < 4; ++j) { accM[i][j] = vzero; accA[i][j] = vzero; }

#pragma unroll 1
  for (int k0 = 0; k0 < NDIM; k0 += BK) {
    __syncthreads();
#pragma unroll
    for (int i = 0; i < 4; ++i) {
      const int f   = t + i * 256;
      const int row = f >> 3;
      const int c4  = f & 7;
      const f4v va = *(const f4v*)(X + (size_t)(rowA0 + row) * NDIM + k0 + c4 * 4);
      const f4v vb = *(const f4v*)(W + (size_t)(rowB0 + row) * NDIM + k0 + c4 * 4);
      unsigned int h, l;
      splitPack(va[0], va[1], h, l);
      sAhi[row * LDSW + c4 * 2 + 0] = h;  sAlo[row * LDSW + c4 * 2 + 0] = l;
      splitPack(va[2], va[3], h, l);
      sAhi[row * LDSW + c4 * 2 + 1] = h;  sAlo[row * LDSW + c4 * 2 + 1] = l;
      splitPack(vb[0], vb[1], h, l);
      sBhi[row * LDSW + c4 * 2 + 0] = h;  sBlo[row * LDSW + c4 * 2 + 0] = l;
      splitPack(vb[2], vb[3], h, l);
      sBhi[row * LDSW + c4 * 2 + 1] = h;  sBlo[row * LDSW + c4 * 2 + 1] = l;
      if (i == 0 && (k0 + BK) < NDIM) {
        __builtin_prefetch((const void*)(X + (size_t)(rowA0 + row) * NDIM + (k0 + BK) + c4 * 4), 0, 3);
        __builtin_prefetch((const void*)(W + (size_t)(rowB0 + row) * NDIM + (k0 + BK) + c4 * 4), 0, 3);
      }
    }
    __syncthreads();

    Frag aH[2], aL[2], bH[4], bL[4], bA[4];
#pragma unroll
    for (int mi = 0; mi < 2; ++mi) {
      const int r = mRow + mi * 16 + fm;
      loadFragA(sAhi, r, kh, aH[mi]);
      loadFragA(sAlo, r, kh, aL[mi]);
    }
#pragma unroll
    for (int ni = 0; ni < 4; ++ni) {
      const int r = nRow + ni * 16 + fm;
      loadFragB(sBhi, r, ks, bH[ni]);
      loadFragB(sBlo, r, ks, bL[ni]);
#pragma unroll
      for (int j = 0; j < 8; ++j) bA[ni].u[j] = bH[ni].u[j] & 0x7FFF7FFFu;
    }
#pragma unroll
    for (int mi = 0; mi < 2; ++mi) {
      Frag aA;
#pragma unroll
      for (int j = 0; j < 8; ++j) aA.u[j] = aH[mi].u[j] & 0x7FFF7FFFu;
#pragma unroll
      for (int ni = 0; ni < 4; ++ni) {
        accM[mi][ni] = WMMA_BF16(aH[mi].v, bH[ni].v, accM[mi][ni]);
        accM[mi][ni] = WMMA_BF16(aH[mi].v, bL[ni].v, accM[mi][ni]);
        accM[mi][ni] = WMMA_BF16(aL[mi].v, bH[ni].v, accM[mi][ni]);
        accA[mi][ni] = WMMA_BF16(aA.v,     bA[ni].v, accA[mi][ni]);
      }
    }
  }

  const int rBase = rowA0 + mRow + (lane >> 4) * 8;
  const int cBase = rowB0 + nRow + (lane & 15);
#pragma unroll
  for (int mi = 0; mi < 2; ++mi) {
#pragma unroll
    for (int ni = 0; ni < 4; ++ni) {
      float* o = Out + (size_t)(rBase + mi * 16) * NDIM + (cBase + ni * 16);
#pragma unroll
      for (int r = 0; r < 8; ++r)
        o[(size_t)r * NDIM] = accM[mi][ni][r] + 0.01f * accA[mi][ni][r] - 1.0e-6f;
    }
  }
}

extern "C" void kernel_launch(void* const* d_in, const int* in_sizes, int n_in,
                              void* d_out, int out_size, void* d_ws, size_t ws_size,
                              hipStream_t stream) {
  (void)in_sizes; (void)n_in; (void)out_size;
  const float* X = (const float*)d_in[0];
  const float* W = (const float*)d_in[1];
  float* Out = (float*)d_out;

  const size_t quarter = (size_t)NDIM * NDIM / 2;          // uints per bf16 matrix
  const size_t need    = 4 * quarter * sizeof(unsigned);   // 128 MiB

  if (ws_size >= need) {
    unsigned int* Ahi = (unsigned int*)d_ws;
    unsigned int* Alo = Ahi + quarter;
    unsigned int* Bhi = Alo + quarter;
    unsigned int* Blo = Bhi + quarter;
    const int cblocks = (NDIM * NDIM / 4) / 256;           // float4 per thread
    split_convert<<<cblocks, 256, 0, stream>>>(X, Ahi, Alo);
    split_convert<<<cblocks, 256, 0, stream>>>(W, Bhi, Blo);
    gemm_bf16_tdm<<<dim3(NDIM / 128, NDIM / 128), 256, 0, stream>>>(Ahi, Alo, Bhi, Blo, Out);
  } else {
    fused_gemm_absgemm<<<dim3(NDIM / 128, NDIM / 128), 256, 0, stream>>>(X, W, Out);
  }
}
